// GraphAwareTensorSketch_21818433864273
// MI455X (gfx1250) — matrix operand order repaired and checked by reading, not compiled, not run on hardware
//
#include <hip/hip_runtime.h>
#include <cstdint>
#include <cstddef>

#define LSEQ   8192
#define KMER   6
#define KSPACE 4096
#define EMB    32
#define NHEADS 4
#define HD     8
#define DSK    64
#define INV_L  (1.0f / 8192.0f)

typedef __attribute__((ext_vector_type(16))) _Float16 v16h;
typedef __attribute__((ext_vector_type(8)))  float    v8f;

union U16 { v16h h; uint4 u[2]; };

// ---------------------------------------------------------------- utilities
__global__ void k_zero(uint32_t* __restrict__ p, size_t n) {
  size_t i = (size_t)blockIdx.x * blockDim.x + threadIdx.x;
  size_t stride = (size_t)gridDim.x * blockDim.x;
  for (; i < n; i += stride) p[i] = 0u;
}

// ------------------------------------------------- sketch: soft hash tables
__global__ void k_prep(const float* __restrict__ hash_logits,
                       const float* __restrict__ sign_logits,
                       float* __restrict__ h_tab, float* __restrict__ s_tab) {
  int t = threadIdx.x;                       // 12 rows: (p,a) flattened
  if (t < 12) {
    const float* r = hash_logits + t * DSK;
    float mx = -3.0e38f;
    for (int d = 0; d < DSK; ++d) mx = fmaxf(mx, r[d]);
    float sum = 0.f;
    for (int d = 0; d < DSK; ++d) sum += __expf(r[d] - mx);
    float inv = 1.0f / sum;
    for (int d = 0; d < DSK; ++d) h_tab[t * DSK + d] = __expf(r[d] - mx) * inv;
    s_tab[t] = tanhf(sign_logits[t]);
  }
}

// level-1: exclusive prefix of s*invL*h  (T[0] is a delta => conv == h row)
__global__ void k_scan1(const int* __restrict__ seq, const float* __restrict__ h_tab,
                        const float* __restrict__ s_tab, float* __restrict__ T1pre) {
  int d = threadIdx.x;                       // 64 threads
  float acc = 0.f;
  for (int t = 0; t < LSEQ; ++t) {
    T1pre[t * DSK + d] = acc;
    int c = seq[t];
    acc += s_tab[c] * INV_L * h_tab[c * DSK + d];
  }
}

// per-t circular conv contribution: val[d] = s*invL * sum_j Tpre[j]*h[(d-j)&63]
__global__ void k_conv(const int* __restrict__ seq, const float* __restrict__ h_tab,
                       const float* __restrict__ s_tab, const float* __restrict__ Tpre,
                       float* __restrict__ outArr, float* __restrict__ baseAcc,
                       int lev, int accumulate) {
  __shared__ float Ts[DSK], Hs[DSK];
  int t = blockIdx.x, d = threadIdx.x;
  int c = seq[t];
  Ts[d] = Tpre[t * DSK + d];
  Hs[d] = h_tab[(lev * 4 + c) * DSK + d];
  __syncthreads();
  float conv = 0.f;
  #pragma unroll
  for (int j = 0; j < DSK; ++j) conv += Ts[j] * Hs[(d - j) & (DSK - 1)];
  float val = s_tab[lev * 4 + c] * INV_L * conv;
  if (accumulate) atomicAdd(&baseAcc[d], val);
  else            outArr[t * DSK + d] = val;
}

__global__ void k_scan_arr(const float* __restrict__ src, float* __restrict__ dst) {
  int d = threadIdx.x;                       // 64 threads, exclusive scan
  float acc = 0.f;
  for (int t = 0; t < LSEQ; ++t) {
    dst[t * DSK + d] = acc;
    acc += src[t * DSK + d];
  }
}

// ---------------------------------------------------------------- presence
__global__ void k_present(const int* __restrict__ seq, float* __restrict__ present) {
  int t = blockIdx.x * blockDim.x + threadIdx.x;
  if (t < LSEQ - KMER + 1) {
    int id = 0;
    #pragma unroll
    for (int i = 0; i < KMER; ++i) id = id * 4 + seq[t + i];
    present[id] = 1.0f;
  }
}

// ------------------------------------------- small dense layers (K x 32@32x32)
__global__ void k_lin32(const float* __restrict__ src, const float* __restrict__ w,
                        const float* __restrict__ b, float* __restrict__ dst, int relu) {
  int idx = blockIdx.x * blockDim.x + threadIdx.x;
  if (idx >= KSPACE * EMB) return;
  int i = idx >> 5, e = idx & 31;
  const float* s = src + (size_t)i * EMB;
  const float* wr = w + e * EMB;
  float a = b[e];
  #pragma unroll
  for (int k = 0; k < EMB; ++k) a += s[k] * wr[k];
  dst[idx] = relu ? fmaxf(a, 0.f) : a;
}

// degree-normalized De Bruijn aggregation: 4 successors per node
__global__ void k_agg(const float* __restrict__ src, const float* __restrict__ present,
                      float* __restrict__ dst) {
  int idx = blockIdx.x * blockDim.x + threadIdx.x;
  if (idx >= KSPACE * EMB) return;
  int i = idx >> 5, e = idx & 31;
  int j0 = (i & 1023) * 4;                   // i % (K/A) == j / A
  float num = 0.f, S = 0.f;
  #pragma unroll
  for (int c = 0; c < 4; ++c) {
    int j = j0 + c;
    float wgt = present[j] * (j != i ? 1.f : 0.f);
    S += wgt;
    num += wgt * src[(size_t)j * EMB + e];
  }
  float pi = present[i];
  dst[idx] = pi * num / (pi * S + 1e-6f);
}

// QKV projection + WMMA-friendly packing (hd=8 zero-padded to K=32)
__global__ void k_qkv(const float* __restrict__ x, const float* __restrict__ w,
                      const float* __restrict__ b, _Float16* __restrict__ q_pack,
                      _Float16* __restrict__ k_pack, _Float16* __restrict__ vT) {
  int idx = blockIdx.x * blockDim.x + threadIdx.x;
  if (idx >= KSPACE * 96) return;
  int i = idx / 96, o = idx % 96;
  const float* xr = x + (size_t)i * EMB;
  const float* wr = w + o * EMB;
  float a = b[o];
  #pragma unroll
  for (int k = 0; k < EMB; ++k) a += xr[k] * wr[k];
  int h = (o & 31) >> 3, d = o & 7;
  if (o < 32)      q_pack[((size_t)(h * KSPACE) + i) * 32 + d] = (_Float16)(a * 0.3535533906f); // fold 1/sqrt(hd)
  else if (o < 64) k_pack[((size_t)(h * KSPACE) + i) * 32 + d] = (_Float16)a;
  else             vT[(size_t)(h * 16 + d) * KSPACE + i] = (_Float16)a;
}

// -------------------------- flash-style masked MHA: one wave per (head, qtile)
__global__ void __launch_bounds__(256)
k_attn(const _Float16* __restrict__ q_pack, const _Float16* __restrict__ k_pack,
       const _Float16* __restrict__ vT, const float* __restrict__ present,
       float* __restrict__ att_pre) {
  __shared__ _Float16 pbuf[8][16 * 32];
  const int lane  = threadIdx.x & 31;
  const int wv    = threadIdx.x >> 5;
  const int gw    = blockIdx.x * 8 + wv;     // 1024 waves total
  const int head  = gw >> 8;
  const int qtile = gw & 255;
  const int lh = lane >> 4;                  // lane half
  const int ln = lane & 15;

  // Q tile, A-matrix layout (16x32 f16): row = lane%16, K chunks per half
  U16 qa;
  {
    const uint4* q4 = (const uint4*)(q_pack + (size_t)(head * KSPACE + qtile * 16 + ln) * 32);
    qa.u[0] = q4[lh];
    qa.u[1] = q4[2 + lh];
  }
  v8f vzero = {0.f, 0.f, 0.f, 0.f, 0.f, 0.f, 0.f, 0.f};

  // ---- pass 1: masked row maxima
  float m8[8];
  #pragma unroll
  for (int i = 0; i < 8; ++i) m8[i] = -3.0e38f;
  for (int kb = 0; kb < KSPACE; kb += 16) {
    U16 kb16;                                // B-matrix 32x16: col = lane%16
    const uint4* k4 = (const uint4*)(k_pack + (size_t)(head * KSPACE + kb + ln) * 32);
    kb16.u[0] = k4[lh]; kb16.u[1] = k4[2 + lh];
    v8f s = __builtin_amdgcn_wmma_f32_16x16x32_f16(false, qa.h, false, kb16.h,
                                                   (short)0, vzero, false, false);
    float pres = present[kb + ln];
    #pragma unroll
    for (int i = 0; i < 8; ++i) {
      float v = pres > 0.f ? s[i] : -1.0e9f;
      m8[i] = fmaxf(m8[i], v);
    }
  }
  #pragma unroll
  for (int off = 1; off < 16; off <<= 1) {
    #pragma unroll
    for (int i = 0; i < 8; ++i) m8[i] = fmaxf(m8[i], __shfl_xor(m8[i], off));
  }

  // ---- pass 2: sum-exp and P@V accumulation (32 keys per PV WMMA)
  float l8[8];
  #pragma unroll
  for (int i = 0; i < 8; ++i) l8[i] = 0.f;
  v8f acc = vzero;
  _Float16* pb = pbuf[wv];

  for (int kb = 0; kb < KSPACE; kb += 32) {
    #pragma unroll
    for (int t = 0; t < 2; ++t) {
      const int k0 = kb + t * 16;
      U16 kb16;
      const uint4* k4 = (const uint4*)(k_pack + (size_t)(head * KSPACE + k0 + ln) * 32);
      kb16.u[0] = k4[lh]; kb16.u[1] = k4[2 + lh];
      v8f s = __builtin_amdgcn_wmma_f32_16x16x32_f16(false, qa.h, false, kb16.h,
                                                     (short)0, vzero, false, false);
      float pres = present[k0 + ln];
      #pragma unroll
      for (int i = 0; i < 8; ++i) {
        float p = pres > 0.f ? __expf(s[i] - m8[i]) : 0.f;
        l8[i] += p;
        pb[(i + 8 * lh) * 32 + (t * 16 + ln)] = (_Float16)p;   // C-layout -> LDS tile
      }
    }
    asm volatile("s_wait_dscnt 0" ::: "memory");               // LDS RAW fence
    U16 pa;                                                    // reload as A-layout
    {
      const uint4* p4 = (const uint4*)(pb + ln * 32);
      pa.u[0] = p4[2 * lh]; pa.u[1] = p4[2 * lh + 1];
    }
    U16 vb;                                                    // V^T as B-matrix
    {
      const uint4* v4 = (const uint4*)(vT + (size_t)(head * 16 + ln) * KSPACE + kb + lh * 16);
      vb.u[0] = v4[0]; vb.u[1] = v4[1];
    }
    acc = __builtin_amdgcn_wmma_f32_16x16x32_f16(false, pa.h, false, vb.h,
                                                 (short)0, acc, false, false);
  }

  #pragma unroll
  for (int off = 1; off < 16; off <<= 1) {
    #pragma unroll
    for (int i = 0; i < 8; ++i) l8[i] += __shfl_xor(l8[i], off);
  }

  if (ln < HD) {
    #pragma unroll
    for (int i = 0; i < 8; ++i) {
      int q = qtile * 16 + i + 8 * lh;
      att_pre[(size_t)q * EMB + head * HD + ln] = acc[i] / l8[i];
    }
  }
}

// ------------------------- output projection + masked mean (atomic reduce)
__global__ void k_attout(const float* __restrict__ att_pre, const float* __restrict__ w,
                         const float* __restrict__ b, const float* __restrict__ present,
                         float* __restrict__ gsum, float* __restrict__ gcnt) {
  int idx = blockIdx.x * blockDim.x + threadIdx.x;
  if (idx >= KSPACE * EMB) return;
  int i = idx >> 5, e = idx & 31;
  const float* ar = att_pre + (size_t)i * EMB;
  const float* wr = w + e * EMB;
  float a = b[e];
  #pragma unroll
  for (int f = 0; f < EMB; ++f) a += ar[f] * wr[f];
  float p = present[i];
  atomicAdd(&gsum[e], p * a);
  if (e == 0) atomicAdd(gcnt, p);
}

// -------------------------------------------- context MLP + fusion head
__global__ void k_final(const float* __restrict__ gsum, const float* __restrict__ gcnt,
                        const float* __restrict__ base,
                        const float* __restrict__ ctx_w1, const float* __restrict__ ctx_b1,
                        const float* __restrict__ ctx_w2, const float* __restrict__ ctx_b2,
                        const float* __restrict__ fus_w1, const float* __restrict__ fus_b1,
                        const float* __restrict__ fus_w2, const float* __restrict__ fus_b2,
                        float* __restrict__ out) {
  __shared__ float mean[32], c1[16], comb[128], f1[128];
  int t = threadIdx.x;                       // 128 threads
  float cnt = gcnt[0];
  if (t < 32) mean[t] = gsum[t] / cnt;
  __syncthreads();
  if (t < 16) {
    float a = ctx_b1[t];
    for (int e = 0; e < 32; ++e) a += ctx_w1[t * 32 + e] * mean[e];
    c1[t] = fmaxf(a, 0.f);
  }
  __syncthreads();
  if (t < 64) {
    comb[t] = base[t];
    float a = ctx_b2[t];
    for (int j = 0; j < 16; ++j) a += ctx_w2[t * 16 + j] * c1[j];
    comb[64 + t] = a;
  }
  __syncthreads();
  {
    float a = fus_b1[t];
    for (int i = 0; i < 128; ++i) a += fus_w1[t * 128 + i] * comb[i];
    f1[t] = fmaxf(a, 0.f);
  }
  __syncthreads();
  if (t < 64) {
    float a = fus_b2[t];
    for (int o = 0; o < 128; ++o) a += fus_w2[t * 128 + o] * f1[o];
    out[t] = tanhf(a);
  }
}

// ---------------------------------------------------------------- launcher
extern "C" void kernel_launch(void* const* d_in, const int* in_sizes, int n_in,
                              void* d_out, int out_size, void* d_ws, size_t ws_size,
                              hipStream_t stream) {
  (void)in_sizes; (void)n_in; (void)out_size; (void)ws_size;
  const int*   seq         = (const int*)  d_in[0];
  const float* hash_logits = (const float*)d_in[1];
  const float* sign_logits = (const float*)d_in[2];
  const float* kmer_emb    = (const float*)d_in[3];
  const float* gc1_w = (const float*)d_in[4];
  const float* gc1_b = (const float*)d_in[5];
  const float* gc2_w = (const float*)d_in[6];
  const float* gc2_b = (const float*)d_in[7];
  const float* attn_in_w  = (const float*)d_in[8];
  const float* attn_in_b  = (const float*)d_in[9];
  const float* attn_out_w = (const float*)d_in[10];
  const float* attn_out_b = (const float*)d_in[11];
  const float* ctx_w1 = (const float*)d_in[12];
  const float* ctx_b1 = (const float*)d_in[13];
  const float* ctx_w2 = (const float*)d_in[14];
  const float* ctx_b2 = (const float*)d_in[15];
  const float* fus_w1 = (const float*)d_in[16];
  const float* fus_b1 = (const float*)d_in[17];
  const float* fus_w2 = (const float*)d_in[18];
  const float* fus_b2 = (const float*)d_in[19];
  float* out = (float*)d_out;

  char* wsp = (char*)d_ws;
  size_t off = 0;
  auto alloc = [&](size_t bytes) -> char* {
    char* p = wsp + off;
    off = (off + bytes + 255) & ~(size_t)255;
    return p;
  };
  // zero-initialized region first (presence, accumulators, padded f16 packs)
  float*    present = (float*)   alloc(KSPACE * 4);
  float*    base    = (float*)   alloc(DSK * 4);
  float*    gsum    = (float*)   alloc(EMB * 4);
  float*    gcnt    = (float*)   alloc(16);
  _Float16* q_pack  = (_Float16*)alloc((size_t)NHEADS * KSPACE * 32 * 2);
  _Float16* k_pack  = (_Float16*)alloc((size_t)NHEADS * KSPACE * 32 * 2);
  _Float16* vT      = (_Float16*)alloc((size_t)NHEADS * 16 * KSPACE * 2);
  size_t zero_bytes = off;
  // plain scratch
  float* h_tab   = (float*)alloc(12 * DSK * 4);
  float* s_tab   = (float*)alloc(64);
  float* T1pre   = (float*)alloc((size_t)LSEQ * DSK * 4);
  float* c2      = (float*)alloc((size_t)LSEQ * DSK * 4);
  float* T2pre   = (float*)alloc((size_t)LSEQ * DSK * 4);
  float* h1      = (float*)alloc((size_t)KSPACE * EMB * 4);
  float* agg1    = (float*)alloc((size_t)KSPACE * EMB * 4);
  float* h2      = (float*)alloc((size_t)KSPACE * EMB * 4);
  float* xbuf    = (float*)alloc((size_t)KSPACE * EMB * 4);
  float* att_pre = (float*)alloc((size_t)KSPACE * EMB * 4);

  k_zero<<<512, 256, 0, stream>>>((uint32_t*)wsp, zero_bytes / 4);

  // tensor sketch (parallelized scan decomposition)
  k_prep<<<1, 32, 0, stream>>>(hash_logits, sign_logits, h_tab, s_tab);
  k_scan1<<<1, 64, 0, stream>>>(seq, h_tab, s_tab, T1pre);
  k_conv<<<LSEQ, 64, 0, stream>>>(seq, h_tab, s_tab, T1pre, c2, base, 1, 0);
  k_scan_arr<<<1, 64, 0, stream>>>(c2, T2pre);
  k_conv<<<LSEQ, 64, 0, stream>>>(seq, h_tab, s_tab, T2pre, c2, base, 2, 1);

  // graph branch
  k_present<<<(LSEQ + 255) / 256, 256, 0, stream>>>(seq, present);
  const int gKE = (KSPACE * EMB + 255) / 256;
  k_lin32<<<gKE, 256, 0, stream>>>(kmer_emb, gc1_w, gc1_b, h1, 1);
  k_agg  <<<gKE, 256, 0, stream>>>(h1, present, agg1);
  k_lin32<<<gKE, 256, 0, stream>>>(agg1, gc2_w, gc2_b, h2, 1);
  k_agg  <<<gKE, 256, 0, stream>>>(h2, present, xbuf);
  k_qkv  <<<(KSPACE * 96 + 255) / 256, 256, 0, stream>>>(xbuf, attn_in_w, attn_in_b,
                                                         q_pack, k_pack, vT);
  k_attn <<<(NHEADS * 256) / 8, 256, 0, stream>>>(q_pack, k_pack, vT, present, att_pre);
  k_attout<<<gKE, 256, 0, stream>>>(att_pre, attn_out_w, attn_out_b, present, gsum, gcnt);

  // heads
  k_final<<<1, 128, 0, stream>>>(gsum, gcnt, base, ctx_w1, ctx_b1, ctx_w2, ctx_b2,
                                 fus_w1, fus_b1, fus_w2, fus_b2, out);
}
